// Quantizer_16415365005665
// MI455X (gfx1250) — matrix operand order repaired and checked by reading, not compiled, not run on hardware
//
#include <hip/hip_runtime.h>

// VQ-VAE quantizer forward, fused for gfx1250 (MI455X).
//   prep kernel: codebook/conv_w -> bf16 + codebook norms into d_ws (once).
//   main kernel: GEMM1 (1x1 conv) + GEMM2 (codebook distances) via
//   v_wmma_f32_16x16x32_bf16, argmin + loss + straight-through output.
// B-operand tiles (x, z) stored pre-paired in LDS as uint32 (bf16 rows 2k,2k+1
// packed per column) -> B fragments are 8x ds_load_b32 with immediate offsets.

#define PT        128   // pixels per workgroup tile
#define NTHREADS  256   // 8 wave32
#define C_IN      128
#define D_DIM     64
#define K_CODES   512
#define HW        4096  // 64*64
#define NPIX_ALL  (32 * HW)
#define NELEM_OUT ((float)(32 * D_DIM * HW))

// d_ws layout (bytes):
//   [0]        loss accumulator (f32)
//   [64]       bf16 codebook   [512][64]  (65536 B)
//   [65600]    bf16 conv_w     [64][128]  (16384 B)
//   [81984]    f32 codebook norms [512]   (2048 B)
#define WS_CB_OFF   64
#define WS_W_OFF    (WS_CB_OFF + K_CODES * D_DIM * 2)
#define WS_C2_OFF   (WS_W_OFF + D_DIM * C_IN * 2)

typedef __attribute__((ext_vector_type(16))) __bf16        v16bf;
typedef __attribute__((ext_vector_type(8)))  float         v8f;
typedef __attribute__((ext_vector_type(8)))  unsigned int  v8u;

__device__ __forceinline__ unsigned int f2bf(float f) {
  unsigned int u = __float_as_uint(f);
  u += 0x7FFFu + ((u >> 16) & 1u);   // round-to-nearest-even
  return u >> 16;
}

__device__ __forceinline__ v8f wmma_bf16(v8u a, v8u b, v8f c) {
  return __builtin_amdgcn_wmma_f32_16x16x32_bf16(
      false, __builtin_bit_cast(v16bf, a),
      false, __builtin_bit_cast(v16bf, b),
      (short)0, c, false, false);
}

// ---- prep: block-invariant conversions, done once ----
__global__ void __launch_bounds__(NTHREADS)
vq_prep_kernel(const float* __restrict__ convw, const float* __restrict__ cb,
               unsigned char* __restrict__ ws) {
  unsigned short* cbB = (unsigned short*)(ws + WS_CB_OFF);
  unsigned short* wB  = (unsigned short*)(ws + WS_W_OFF);
  float*          c2g = (float*)(ws + WS_C2_OFF);
  const int gid = blockIdx.x * NTHREADS + threadIdx.x;
  const int gsz = gridDim.x * NTHREADS;
  for (int t = gid; t < K_CODES * D_DIM; t += gsz) cbB[t] = (unsigned short)f2bf(cb[t]);
  for (int t = gid; t < D_DIM * C_IN; t += gsz)    wB[t]  = (unsigned short)f2bf(convw[t]);
  for (int k = gid; k < K_CODES; k += gsz) {
    const float* row = cb + k * D_DIM;
    float s = 0.f;
    #pragma unroll 8
    for (int d = 0; d < D_DIM; ++d) { float v = row[d]; s += v * v; }
    c2g[k] = s;
  }
  if (gid == 0) *(float*)ws = 0.f;   // zero loss accumulator
}

__global__ void __launch_bounds__(NTHREADS)
vq_fused_kernel(const float* __restrict__ x, const float* __restrict__ convb,
                const float* __restrict__ cb, const unsigned char* __restrict__ ws_ro,
                float* __restrict__ out, float* __restrict__ lossws) {
  extern __shared__ __align__(16) unsigned char smem_raw[];
  unsigned short* wA   = (unsigned short*)smem_raw;        // [64][128]    bf16 conv_w (A op)
  unsigned short* cbT  = wA + D_DIM * C_IN;                // [512][64]    bf16 codebook (A op)
  unsigned int*   xTp  = (unsigned int*)(cbT + K_CODES * D_DIM); // [64][PT]  packed x (B op)
  unsigned int*   zTp  = xTp + (C_IN / 2) * PT;            // [32][PT]     packed z (B op)
  float* c2    = (float*)(zTp + (D_DIM / 2) * PT);         // [512] codebook sq norms
  float* biasL = c2 + K_CODES;                             // [64]
  float* z2L   = biasL + D_DIM;                            // [PT] per-pixel |z|^2
  float* minvL = z2L + PT;                                 // [PT] per-pixel min dist (minus z^2)
  int*   idxL  = (int*)(minvL + PT);                       // [PT] argmin code
  float* lossL = (float*)(idxL + PT);                      // [1]

  const int tid     = threadIdx.x;
  const int pixbase = blockIdx.x * PT;
  const int b       = pixbase / HW;
  const int pixoff  = pixbase % HW;

  // ---- stage pre-converted bf16 operands with b128 copies ----
  {
    const uint4* s1 = (const uint4*)(ws_ro + WS_CB_OFF);
    uint4*       d1 = (uint4*)cbT;
    for (int t = tid; t < (K_CODES * D_DIM) / 8; t += NTHREADS) d1[t] = s1[t];
    const uint4* s2 = (const uint4*)(ws_ro + WS_W_OFF);
    uint4*       d2 = (uint4*)wA;
    for (int t = tid; t < (D_DIM * C_IN) / 8; t += NTHREADS) d2[t] = s2[t];
    const float* s3 = (const float*)(ws_ro + WS_C2_OFF);
    for (int t = tid; t < K_CODES; t += NTHREADS) c2[t] = s3[t];
  }
  if (tid < D_DIM) biasL[tid] = convb[tid];
  if (tid < PT)    z2L[tid]   = 0.f;
  if (tid == 0)    lossL[0]   = 0.f;
  // ---- stage x tile: f32 -> bf16, pre-paired for the B-operand layout ----
  const float* xsrc = x + ((size_t)b * C_IN) * HW + pixoff;
  for (int t = tid; t < (C_IN / 2) * PT; t += NTHREADS) {
    int cp = t >> 7;          // channel pair
    int j  = t & (PT - 1);
    unsigned int lo = f2bf(xsrc[(size_t)(2 * cp) * HW + j]);
    unsigned int hi = f2bf(xsrc[(size_t)(2 * cp + 1) * HW + j]);
    xTp[cp * PT + j] = lo | (hi << 16);   // K=2cp in [15:0], 2cp+1 in [31:16]
  }
  __syncthreads();

  const int wave = tid >> 5;
  const int lane = tid & 31;
  const int h    = lane >> 4;     // half-wave
  const int ln   = lane & 15;
  const int n0   = wave * 16;     // this wave's 16-pixel column stripe

  // per-lane base pointers: per-v offsets become ds immediate offsets
  const unsigned int* xB = xTp + 8 * h * PT + n0 + ln;
  const unsigned int* zB = zTp + 8 * h * PT + n0 + ln;
  unsigned int*       zW = zTp + n0 + ln;

  // ---- GEMM1: z[64 x PT] = conv_w[64x128] @ x[128 x PT] ----
  v8f acc[4] = {};
  #pragma unroll
  for (int ks = 0; ks < 4; ++ks) {
    const int kb = ks * 32;
    v8u bfrg;
    #pragma unroll
    for (int v = 0; v < 8; ++v)           // B layout: K = kb + 16h + 2v (pair pre-packed)
      bfrg[v] = xB[(ks * 16 + v) * PT];
    #pragma unroll
    for (int mt = 0; mt < 4; ++mt) {
      const unsigned short* arow = wA + (mt * 16 + ln) * C_IN + kb + 16 * h;
      v8u afrg;
      #pragma unroll
      for (int v = 0; v < 8; ++v)
        afrg[v] = *(const unsigned int*)(arow + 2 * v);
      acc[mt] = wmma_bf16(afrg, bfrg, acc[mt]);
    }
  }
  // epilogue: bias, |z|^2, z -> LDS packed bf16 (wave-private pixel stripe; no barrier)
  {
    float zsq = 0.f;
    #pragma unroll
    for (int mt = 0; mt < 4; ++mt) {
      #pragma unroll
      for (int v = 0; v < 8; v += 2) {
        int d0 = mt * 16 + 8 * h + v;       // C/D layout: M = v + 8*half; d0 even
        float z0 = acc[mt][v]     + biasL[d0];
        float z1 = acc[mt][v + 1] + biasL[d0 + 1];
        zsq += z0 * z0 + z1 * z1;
        zW[(d0 >> 1) * PT] = f2bf(z0) | (f2bf(z1) << 16);
      }
    }
    atomicAdd(&z2L[n0 + ln], zsq);          // ds_add_f32; two lanes cover all 64 d
  }

  // ---- GEMM2: s[512 x PT] = codebook[512x64] @ z[64 x PT]; argmin_k c2[k]-2s ----
  v8u bz[2];
  #pragma unroll
  for (int ks = 0; ks < 2; ++ks)
    #pragma unroll
    for (int v = 0; v < 8; ++v)
      bz[ks][v] = zB[(ks * 16 + v) * PT];

  float minv = 3.4e38f;
  int   mink = 0;
  for (int mt = 0; mt < 32; ++mt) {
    v8f s = {};
    const unsigned short* arow = cbT + (mt * 16 + ln) * D_DIM + 16 * h;
    #pragma unroll
    for (int ks = 0; ks < 2; ++ks) {
      v8u afrg;
      #pragma unroll
      for (int v = 0; v < 8; ++v)
        afrg[v] = *(const unsigned int*)(arow + ks * 32 + 2 * v);
      s = wmma_bf16(afrg, bz[ks], s);
    }
    #pragma unroll
    for (int v = 0; v < 8; ++v) {
      int k = mt * 16 + 8 * h + v;
      float dist = c2[k] - 2.0f * s[v];     // + |z|^2 is pixel-constant: irrelevant to argmin
      if (dist < minv) { minv = dist; mink = k; }
    }
  }
  // merge half-waves: lanes L and L+16 hold the same pixel column
  {
    float om = __shfl_xor(minv, 16, 32);
    int   ok = __shfl_xor(mink, 16, 32);
    if (om < minv) { minv = om; mink = ok; }
    if (h == 0) { minvL[n0 + ln] = minv; idxL[n0 + ln] = mink; }
  }
  __syncthreads();

  // ---- loss: sum_p (minDist + |z|^2) ; reduce block -> one global atomic ----
  if (tid < PT) atomicAdd(lossL, minvL[tid] + z2L[tid]);

  // ---- quantized_st output == codebook[idx] exactly (f32, L2-hot gather) ----
  float* obase = out + ((size_t)b * D_DIM) * HW + pixoff;
  for (int t = tid; t < D_DIM * PT; t += NTHREADS) {
    int d = t >> 7;
    int j = t & (PT - 1);
    obase[(size_t)d * HW + j] = cb[idxL[j] * D_DIM + d];
  }
  __syncthreads();
  if (tid == 0) atomicAdd(lossws, lossL[0]);
}

__global__ void vq_finalize(const float* ws, float* loss_out) {
  // loss = commitment(0.25) + embedding(1.0) = 1.25 * mean((q - z)^2)
  if (threadIdx.x == 0) loss_out[0] = 1.25f * ws[0] / NELEM_OUT;
}

extern "C" void kernel_launch(void* const* d_in, const int* in_sizes, int n_in,
                              void* d_out, int out_size, void* d_ws, size_t ws_size,
                              hipStream_t stream) {
  (void)in_sizes; (void)n_in; (void)out_size; (void)ws_size;
  const float* x  = (const float*)d_in[0];
  const float* w  = (const float*)d_in[1];
  const float* bb = (const float*)d_in[2];
  const float* cb = (const float*)d_in[3];
  float* out = (float*)d_out;
  unsigned char* ws = (unsigned char*)d_ws;

  vq_prep_kernel<<<64, NTHREADS, 0, stream>>>(w, cb, ws);

  size_t smem =
      (size_t)(D_DIM * C_IN + K_CODES * D_DIM) * sizeof(unsigned short)     // A operands
      + (size_t)((C_IN / 2) * PT + (D_DIM / 2) * PT) * sizeof(unsigned int) // packed B operands
      + (size_t)(K_CODES + D_DIM + PT + PT) * sizeof(float)
      + (size_t)PT * sizeof(int) + 16;     // ~132 KB of the 320 KB WGP LDS

  int nblocks = NPIX_ALL / PT;             // 1024
  vq_fused_kernel<<<nblocks, NTHREADS, smem, stream>>>(x, bb, cb, ws, out, (float*)ws);

  vq_finalize<<<1, 32, 0, stream>>>((const float*)ws, out + (size_t)32 * D_DIM * HW);
}
